// GCNSY_62783831933365
// MI455X (gfx1250) — compile-verified, matrix operand order checked by reading, not run on hardware
//
#include <hip/hip_runtime.h>
#include <hip/hip_bf16.h>
#include <math.h>

typedef __attribute__((ext_vector_type(16))) __bf16 v16bf;
typedef __attribute__((ext_vector_type(8)))  __bf16 v8bf;
typedef __attribute__((ext_vector_type(4)))  __bf16 v4bf;
typedef __attribute__((ext_vector_type(8)))  float  v8f;

#define HDIM 256

// ---------------------------------------------------------------------------
// Fragment layout (CDNA5 WMMA 16-bit A/B, ISA 7.12.2), per lane:
//   elems 0..7  = K = k0 + half*8 + {0..7}
//   elems 8..15 = K = k0 + 16 + half*8 + {0..7}
// ---------------------------------------------------------------------------
__device__ __forceinline__ v16bf frag_from_bf16(const __bf16* __restrict__ p) {
  v8bf lo = *reinterpret_cast<const v8bf*>(p);        // 16B aligned
  v8bf hi = *reinterpret_cast<const v8bf*>(p + 16);
  v16bf f;
#pragma unroll
  for (int i = 0; i < 8; ++i) { f[i] = lo[i]; f[8 + i] = hi[i]; }
  return f;
}

__device__ __forceinline__ v16bf frag_from_f32_fast(const float* __restrict__ p) {
  v16bf f;
#pragma unroll
  for (int i = 0; i < 8; ++i) f[i]     = (__bf16)p[i];
#pragma unroll
  for (int i = 0; i < 8; ++i) f[8 + i] = (__bf16)p[16 + i];
  return f;
}

__device__ __forceinline__ v16bf frag_from_f32_tail(const float* __restrict__ p,
                                                    int k, int K) {
  v16bf f;
#pragma unroll
  for (int i = 0; i < 8; ++i) f[i]     = (k + i      < K) ? (__bf16)p[i]      : (__bf16)0.0f;
#pragma unroll
  for (int i = 0; i < 8; ++i) f[8 + i] = (k + 16 + i < K) ? (__bf16)p[16 + i] : (__bf16)0.0f;
  return f;
}

#define WMMA_BF16(A, B, C) \
  __builtin_amdgcn_wmma_f32_16x16x32_bf16(false, (A), false, (B), (short)0, (C), false, false)

// ---------------------------------------------------------------------------
// Zero-padded fp32 -> bf16 weight conversion: wb[H x Kpad] from w[H x K].
// ---------------------------------------------------------------------------
__global__ __launch_bounds__(256) void convert_weight_bf16(
    const float* __restrict__ w, __bf16* __restrict__ wb, int K, int Kpad, int total) {
  int i = blockIdx.x * 256 + threadIdx.x;
  if (i >= total) return;
  int r = i / Kpad, c = i - r * Kpad;
  wb[i] = (c < K) ? (__bf16)w[(size_t)r * K + c] : (__bf16)0.0f;
}

// ---------------------------------------------------------------------------
// WMMA GEMM: C[M x 256] = A[M x K] * W^T.  Wb is [256 x Kpad] bf16 (padded).
// Block = 256 threads = 8 waves: 2 waves over M (32 rows = 2 WMMA tiles each),
// 4 over N (64 cols = 4 WMMA tiles).  Block tile = 64 x 256.
// Each B fragment feeds 2 WMMAs (register blocking over M).
// MODE 0: GCN -> h_out = C ; agg_out = C*dinv[row]^2 + bias[col]
// MODE 1: ELU -> xnext(bf16) = elu(C + bias)
// MODE 2: OUT -> final_out(fp32) = C + bias
// ---------------------------------------------------------------------------
template <int MODE, bool AF32>
__global__ __launch_bounds__(256) void gemm_wmma(
    const void* __restrict__ Av, const __bf16* __restrict__ Wb,
    int M, int K, int Kpad,
    const float* __restrict__ bias, const float* __restrict__ dinv,
    float* __restrict__ h_out, float* __restrict__ agg_out,
    __bf16* __restrict__ xnext, float* __restrict__ final_out) {
  const int tid  = threadIdx.x;
  const int lane = tid & 31;
  const int wid  = tid >> 5;
  const int half = lane >> 4;
  const int l16  = lane & 15;

  const int mBase = blockIdx.x * 64 + (wid & 1) * 32;   // 2 M-tiles: mBase, mBase+16
  const int nBase = (wid >> 1) * 64;
  const bool full = (int)(blockIdx.x + 1) * 64 <= M;    // uniform per block

  const int aRow0 = min(mBase + l16,      M - 1);
  const int aRow1 = min(mBase + 16 + l16, M - 1);
  const int kOff  = half * 8;

  // streaming pointers (advance by 32 elements per k-step)
  const float*  apf0 = (const float*)Av  + (size_t)aRow0 * K + kOff;
  const float*  apf1 = (const float*)Av  + (size_t)aRow1 * K + kOff;
  const __bf16* apb0 = (const __bf16*)Av + (size_t)aRow0 * Kpad + kOff;
  const __bf16* apb1 = (const __bf16*)Av + (size_t)aRow1 * Kpad + kOff;
  const __bf16* bp0  = Wb + (size_t)(nBase +  0 + l16) * Kpad + kOff;
  const __bf16* bp1  = Wb + (size_t)(nBase + 16 + l16) * Kpad + kOff;
  const __bf16* bp2  = Wb + (size_t)(nBase + 32 + l16) * Kpad + kOff;
  const __bf16* bp3  = Wb + (size_t)(nBase + 48 + l16) * Kpad + kOff;

  v8f acc00 = {}, acc01 = {}, acc02 = {}, acc03 = {};   // M-tile 0, N-tiles 0..3
  v8f acc10 = {}, acc11 = {}, acc12 = {}, acc13 = {};   // M-tile 1, N-tiles 0..3

  const int kIter = K >> 5;                 // branch-free main loop
  for (int it = 0; it < kIter; ++it) {
    v16bf a0 = AF32 ? frag_from_f32_fast(apf0) : frag_from_bf16(apb0);
    v16bf a1 = AF32 ? frag_from_f32_fast(apf1) : frag_from_bf16(apb1);
    v16bf b0 = frag_from_bf16(bp0);
    v16bf b1 = frag_from_bf16(bp1);
    v16bf b2 = frag_from_bf16(bp2);
    v16bf b3 = frag_from_bf16(bp3);
    acc00 = WMMA_BF16(a0, b0, acc00);
    acc10 = WMMA_BF16(a1, b0, acc10);
    acc01 = WMMA_BF16(a0, b1, acc01);
    acc11 = WMMA_BF16(a1, b1, acc11);
    acc02 = WMMA_BF16(a0, b2, acc02);
    acc12 = WMMA_BF16(a1, b2, acc12);
    acc03 = WMMA_BF16(a0, b3, acc03);
    acc13 = WMMA_BF16(a1, b3, acc13);
    apf0 += 32; apf1 += 32; apb0 += 32; apb1 += 32;
    bp0 += 32; bp1 += 32; bp2 += 32; bp3 += 32;
  }

  if (AF32 && (K & 31)) {                   // single guarded tail step (layer 0)
    const int k0 = K & ~31;
    v16bf a0 = frag_from_f32_tail((const float*)Av + (size_t)aRow0 * K + k0 + kOff,
                                  k0 + kOff, K);
    v16bf a1 = frag_from_f32_tail((const float*)Av + (size_t)aRow1 * K + k0 + kOff,
                                  k0 + kOff, K);
    v16bf b0 = frag_from_bf16(bp0);         // B is zero-padded to Kpad
    v16bf b1 = frag_from_bf16(bp1);
    v16bf b2 = frag_from_bf16(bp2);
    v16bf b3 = frag_from_bf16(bp3);
    acc00 = WMMA_BF16(a0, b0, acc00);
    acc10 = WMMA_BF16(a1, b0, acc10);
    acc01 = WMMA_BF16(a0, b1, acc01);
    acc11 = WMMA_BF16(a1, b1, acc11);
    acc02 = WMMA_BF16(a0, b2, acc02);
    acc12 = WMMA_BF16(a1, b2, acc12);
    acc03 = WMMA_BF16(a0, b3, acc03);
    acc13 = WMMA_BF16(a1, b3, acc13);
  }

  // C/D layout: elem r of acc -> row = mTileBase + r + 8*half, col = nTile + l16
  float di2[2][8];
  if (MODE == 0) {
#pragma unroll
    for (int mt = 0; mt < 2; ++mt)
#pragma unroll
      for (int r = 0; r < 8; ++r) {
        const float d = dinv[min(mBase + mt * 16 + r + half * 8, M - 1)];
        di2[mt][r] = d * d;
      }
  }

  const v8f accs[2][4] = {{acc00, acc01, acc02, acc03},
                          {acc10, acc11, acc12, acc13}};
#pragma unroll
  for (int t = 0; t < 4; ++t) {
    const int col = nBase + t * 16 + l16;
    const float bn = bias[col];
#pragma unroll
    for (int mt = 0; mt < 2; ++mt) {
#pragma unroll
      for (int r = 0; r < 8; ++r) {
        const int row = mBase + mt * 16 + r + half * 8;
        if (full || row < M) {
          const float v = accs[mt][t][r];
          const size_t o = (size_t)row * HDIM + col;
          if (MODE == 0) {
            h_out[o]   = v;
            agg_out[o] = v * di2[mt][r] + bn;  // self-loop + bias folded in
          } else if (MODE == 1) {
            float u = v + bn;
            u = (u > 0.0f) ? u : (__expf(u) - 1.0f);
            xnext[o] = (__bf16)u;
          } else {
            final_out[o] = v + bn;
          }
        }
      }
    }
  }
}

// ---------------------------------------------------------------------------
// Edge scatter: agg[dst] += h[src] * dinv[src]*dinv[dst].  4 edges / block,
// 64 threads per edge, float4 per thread; atomics resolve in the 192MB L2.
// ---------------------------------------------------------------------------
__global__ __launch_bounds__(256) void edge_scatter(
    const int* __restrict__ src, const int* __restrict__ dst,
    const float* __restrict__ dinv, const float* __restrict__ h,
    float* __restrict__ agg, int E) {
  const int e = blockIdx.x * 4 + (threadIdx.x >> 6);
  if (e >= E) return;
  const int s = src[e], d = dst[e];
  const float coef = dinv[s] * dinv[d];
  const int f = (threadIdx.x & 63) * 4;
  const float4 hv = *reinterpret_cast<const float4*>(h + (size_t)s * HDIM + f);
  float* ap = agg + (size_t)d * HDIM + f;
  atomicAdd(ap + 0, hv.x * coef);
  atomicAdd(ap + 1, hv.y * coef);
  atomicAdd(ap + 2, hv.z * coef);
  atomicAdd(ap + 3, hv.w * coef);
}

// ---------------------------------------------------------------------------
// relu + L2 row-normalize, emit bf16 for next WMMA layer. One wave32 per row.
// ---------------------------------------------------------------------------
__global__ __launch_bounds__(256) void relu_l2norm_bf16(
    const float* __restrict__ agg, __bf16* __restrict__ xn, int M) {
  const int lane = threadIdx.x & 31;
  const int row  = blockIdx.x * 8 + (threadIdx.x >> 5);
  if (row >= M) return;
  const float4* p = reinterpret_cast<const float4*>(agg + (size_t)row * HDIM);
  float4 v0 = p[lane];
  float4 v1 = p[lane + 32];
  v0.x = fmaxf(v0.x, 0.f); v0.y = fmaxf(v0.y, 0.f);
  v0.z = fmaxf(v0.z, 0.f); v0.w = fmaxf(v0.w, 0.f);
  v1.x = fmaxf(v1.x, 0.f); v1.y = fmaxf(v1.y, 0.f);
  v1.z = fmaxf(v1.z, 0.f); v1.w = fmaxf(v1.w, 0.f);
  float s = v0.x * v0.x + v0.y * v0.y + v0.z * v0.z + v0.w * v0.w +
            v1.x * v1.x + v1.y * v1.y + v1.z * v1.z + v1.w * v1.w;
#pragma unroll
  for (int m = 16; m >= 1; m >>= 1) s += __shfl_xor(s, m, 32);
  const float scale = 1.0f / fmaxf(sqrtf(s), 1e-12f);
  v4bf o0, o1;
  o0[0] = (__bf16)(v0.x * scale); o0[1] = (__bf16)(v0.y * scale);
  o0[2] = (__bf16)(v0.z * scale); o0[3] = (__bf16)(v0.w * scale);
  o1[0] = (__bf16)(v1.x * scale); o1[1] = (__bf16)(v1.y * scale);
  o1[2] = (__bf16)(v1.z * scale); o1[3] = (__bf16)(v1.w * scale);
  *reinterpret_cast<v4bf*>(xn + (size_t)row * HDIM + lane * 4)        = o0;
  *reinterpret_cast<v4bf*>(xn + (size_t)row * HDIM + (lane + 32) * 4) = o1;
}

// --------------------------- degree / dinv ---------------------------------
__global__ void dinv_init(float* deg, int M) {
  int i = blockIdx.x * 256 + threadIdx.x;
  if (i < M) deg[i] = 1.0f;                 // self-loop
}
__global__ void dinv_count(const int* __restrict__ dst, float* deg, int E) {
  int e = blockIdx.x * 256 + threadIdx.x;
  if (e < E) atomicAdd(&deg[dst[e]], 1.0f);
}
__global__ void dinv_finish(float* deg, int M) {
  int i = blockIdx.x * 256 + threadIdx.x;
  if (i < M) deg[i] = rsqrtf(deg[i]);
}

// ---------------------------------------------------------------------------
extern "C" void kernel_launch(void* const* d_in, const int* in_sizes, int n_in,
                              void* d_out, int out_size, void* d_ws, size_t ws_size,
                              hipStream_t stream) {
  const float* x     = (const float*)d_in[0];
  const int*   edges = (const int*)d_in[1];
  const float* Wg0   = (const float*)d_in[2];
  const float* bg0   = (const float*)d_in[3];
  const float* Wg1   = (const float*)d_in[4];
  const float* bg1   = (const float*)d_in[5];
  const float* Wg2   = (const float*)d_in[6];
  const float* bg2   = (const float*)d_in[7];
  const float* W1    = (const float*)d_in[8];
  const float* b1    = (const float*)d_in[9];
  const float* W2    = (const float*)d_in[10];
  const float* b2    = (const float*)d_in[11];

  const int H   = HDIM;
  const int IN  = in_sizes[2] / H;          // Wg0 is [H, IN]
  const int N   = in_sizes[0] / IN;
  const int E   = in_sizes[1] / 2;
  const int KP0 = (IN + 31) & ~31;          // padded K for layer 0
  const int* src = edges;
  const int* dst = edges + E;

  char* ws = (char*)d_ws;
  size_t off = 0;
  auto alloc = [&](size_t bytes) -> void* {
    void* p = ws + off;
    off += (bytes + 255) & ~(size_t)255;
    return p;
  };
  float*  dinv = (float*)alloc((size_t)N * 4);
  float*  h    = (float*)alloc((size_t)N * H * 4);
  float*  agg  = (float*)alloc((size_t)N * H * 4);
  __bf16* xb1  = (__bf16*)alloc((size_t)N * H * 2);
  __bf16* xb2  = (__bf16*)alloc((size_t)N * H * 2);
  __bf16* wg0b = (__bf16*)alloc((size_t)H * KP0 * 2);
  __bf16* wg1b = (__bf16*)alloc((size_t)H * H * 2);
  __bf16* wg2b = (__bf16*)alloc((size_t)H * H * 2);
  __bf16* w1b  = (__bf16*)alloc((size_t)H * H * 2);
  __bf16* w2b  = (__bf16*)alloc((size_t)H * H * 2);

  dim3 blk(256);
  const int ng = (N + 255) / 256;
  const int eg = (E + 255) / 256;
  const int gg = (N + 63) / 64;             // GEMM grid (64 rows / block)
  const int sg = (E + 3) / 4;               // scatter grid (4 edges / block)
  const int rg = (N + 7) / 8;               // norm grid (8 rows / block)
  const int w0g = (H * KP0 + 255) / 256;
  const int whg = (H * H + 255) / 256;

  // Weight preconversion to zero-padded bf16 (tiny, once per call)
  convert_weight_bf16<<<w0g, blk, 0, stream>>>(Wg0, wg0b, IN, KP0, H * KP0);
  convert_weight_bf16<<<whg, blk, 0, stream>>>(Wg1, wg1b, H, H, H * H);
  convert_weight_bf16<<<whg, blk, 0, stream>>>(Wg2, wg2b, H, H, H * H);
  convert_weight_bf16<<<whg, blk, 0, stream>>>(W1,  w1b,  H, H, H * H);
  convert_weight_bf16<<<whg, blk, 0, stream>>>(W2,  w2b,  H, H, H * H);

  // dinv = rsqrt(deg + 1)
  dinv_init  <<<ng, blk, 0, stream>>>(dinv, N);
  dinv_count <<<eg, blk, 0, stream>>>(dst, dinv, E);
  dinv_finish<<<ng, blk, 0, stream>>>(dinv, N);

  // GCN layer 0 (fp32 A streamed once, converted in-register; K tail guarded)
  gemm_wmma<0, true ><<<gg, blk, 0, stream>>>(x,   wg0b, N, IN, KP0, bg0, dinv, h, agg, nullptr, nullptr);
  edge_scatter     <<<sg, blk, 0, stream>>>(src, dst, dinv, h, agg, E);
  relu_l2norm_bf16 <<<rg, blk, 0, stream>>>(agg, xb1, N);

  // GCN layer 1 (bf16 A, K = 256)
  gemm_wmma<0, false><<<gg, blk, 0, stream>>>(xb1, wg1b, N, H, H, bg1, dinv, h, agg, nullptr, nullptr);
  edge_scatter     <<<sg, blk, 0, stream>>>(src, dst, dinv, h, agg, E);
  relu_l2norm_bf16 <<<rg, blk, 0, stream>>>(agg, xb2, N);

  // GCN layer 2
  gemm_wmma<0, false><<<gg, blk, 0, stream>>>(xb2, wg2b, N, H, H, bg2, dinv, h, agg, nullptr, nullptr);
  edge_scatter     <<<sg, blk, 0, stream>>>(src, dst, dinv, h, agg, E);
  relu_l2norm_bf16 <<<rg, blk, 0, stream>>>(agg, xb1, N);

  // MLP: ELU(x W1^T + b1) -> bf16
  gemm_wmma<1, false><<<gg, blk, 0, stream>>>(xb1, w1b, N, H, H, b1, nullptr, nullptr, nullptr, xb2, nullptr);
  // Final: x W2^T + b2 -> fp32 output
  gemm_wmma<2, false><<<gg, blk, 0, stream>>>(xb2, w2b, N, H, H, b2, nullptr, nullptr, nullptr, nullptr, (float*)d_out);
}
